// PhysicsInformedHardProj_89300960018671
// MI455X (gfx1250) — matrix operand order repaired and checked by reading, not compiled, not run on hardware
//
#include <hip/hip_runtime.h>

// CDNA5 / gfx1250: wave32, WMMA f16 -> f32 16x16x32, shuffle-based transposes.
typedef _Float16 f16;
typedef __attribute__((ext_vector_type(16))) _Float16 v16h;
typedef __attribute__((ext_vector_type(8)))  _Float16 v8h;
typedef __attribute__((ext_vector_type(2)))  _Float16 v2h;
typedef __attribute__((ext_vector_type(8)))  float    v8f;
typedef __attribute__((ext_vector_type(4)))  float    v4f;

#define WAVES_PER_BLOCK 8
#define BLOCK_THREADS   256
#define GRID_BLOCKS     1024

__device__ __forceinline__ v8f wmma16(v16h a, v16h b, v8f c) {
  // (neg_a, A, neg_b, B, c_mod, C, reuse_a, reuse_b)
  return __builtin_amdgcn_wmma_f32_16x16x32_f16(false, a, false, b, (short)0, c,
                                                false, false);
}

// A-fragment (16xK chunk) for one lane from a 64-f16 row in LDS:
// VGPRs 0-3 hold K = 32*kc + 16*hi + [0,8), VGPRs 4-7 hold +8 more.
__device__ __forceinline__ v16h load_frag(const f16* row, int kc, int hi) {
  const int base = 32 * kc + 16 * hi;
  v8h lo = *(const v8h*)(row + base);
  v8h hh = *(const v8h*)(row + base + 8);
  v16h r;
#pragma unroll
  for (int i = 0; i < 8; ++i) { r[i] = lo[i]; r[i + 8] = hh[i]; }
  return r;
}

__device__ __forceinline__ unsigned pack2(float a, float b) {
  union { v2h h; unsigned u; } t;
  t.h[0] = (f16)a; t.h[1] = (f16)b;
  return t.u;  // v_cvt_pk_f16_f32
}

__device__ __forceinline__ v16h assemble(const unsigned* a, const unsigned* b) {
  union { v16h h; unsigned u[8]; } r;
#pragma unroll
  for (int i = 0; i < 4; ++i) { r.u[i] = a[i]; r.u[4 + i] = b[i]; }
  return r.h;
}

__global__ __launch_bounds__(BLOCK_THREADS)
void mlp_hardproj_kernel(const float* __restrict__ pc,
                         const float* __restrict__ W1,  const float* __restrict__ b1,
                         const float* __restrict__ W2,  const float* __restrict__ b2,
                         const float* __restrict__ W3,  const float* __restrict__ b3,
                         const float* __restrict__ Wmu, const float* __restrict__ bmu,
                         const float* __restrict__ Wlam,const float* __restrict__ blam,
                         float* __restrict__ out, int N)
{
  // Weights staged transposed (wt[m][k] = W[k][m]) so A-fragments (rows of W^T)
  // are contiguous 16B LDS loads.
  __shared__ f16 w2t[64 * 64];
  __shared__ f16 w3t[64 * 64];
  __shared__ f16 whh[16 * 64];      // [Wmu | Wlam | 0-pad]^T : 16 head rows
  __shared__ float w1s[128], b1s[64], b2s[64], b3s[64];

  const int tid = threadIdx.x;

  // ---- one-time staging ----
  for (int i = tid; i < 64 * 64; i += BLOCK_THREADS) {
    int k = i >> 6, n = i & 63;            // W row-major: i = k*64 + n
    w2t[n * 64 + k] = (f16)W2[i];
    w3t[n * 64 + k] = (f16)W3[i];
  }
  for (int i = tid; i < 16 * 64; i += BLOCK_THREADS) {
    int m = i >> 6, k = i & 63;            // head m, feature k
    float v = 0.0f;
    if (m < 4)      v = Wmu[k * 4 + m];
    else if (m < 7) v = Wlam[k * 3 + (m - 4)];
    whh[i] = (f16)v;
  }
  if (tid < 128) w1s[tid] = W1[tid];
  if (tid < 64) { b1s[tid] = b1[tid]; b2s[tid] = b2[tid]; b3s[tid] = b3[tid]; }
  __syncthreads();

  const int lane = tid & 31;
  const int wave = tid >> 5;
  const int hb   = lane >> 4;       // lane half
  const int l15  = lane & 15;

  // ---- only layer-2 weight fragments stay register-resident (64 VGPRs) ----
  v16h A2[4][2];
#pragma unroll
  for (int mt = 0; mt < 4; ++mt) {
    const f16* r2 = w2t + (16 * mt + l15) * 64;
    A2[mt][0] = load_frag(r2, 0, hb);
    A2[mt][1] = load_frag(r2, 1, hb);
  }

  float hbm[4] = {bmu[0], bmu[1], bmu[2], bmu[3]};
  float hbl[3] = {blam[0], blam[1], blam[2]};

  const int ntiles  = (N + 15) >> 4;
  const int ngroups = (ntiles + WAVES_PER_BLOCK - 1) / WAVES_PER_BLOCK;

  for (int g = blockIdx.x; g < ngroups; g += gridDim.x) {
    const int tile    = g * WAVES_PER_BLOCK + wave;   // wave-uniform
    const bool active = (tile < ntiles);
    const int p0 = (active ? tile : 0) * 16;
    const int pt = p0 + l15;                          // this lane's point

    const int tile2 = (g + (int)gridDim.x) * WAVES_PER_BLOCK + wave;
    if (tile2 < ntiles)
      __builtin_prefetch(pc + 2 * tile2 * 16, 0, 0);  // global_prefetch_b8

    // Opaque zero offset: defeats LICM so A3/Ah fragments are re-loaded from
    // LDS each iteration (keeps live VGPRs < 256) while staying ds_load's.
    int off = 0;
    asm volatile("" : "+s"(off));

    // ---- layer 1 (K=2): VALU straight into B-fragment layout.
    // B frag kc, lane half hb: element e <-> feature 32*kc + 16*hb + e.
    const float x0 = pc[2 * pt];
    const float x1 = pc[2 * pt + 1];
    v16h Bx[2];
#pragma unroll
    for (int kc = 0; kc < 2; ++kc) {
      union { v16h h; unsigned u[8]; } bx;
#pragma unroll
      for (int p = 0; p < 8; ++p) {
        const int j = 32 * kc + 16 * hb + 2 * p;
        float f0 = fmaf(x0, w1s[j],     fmaf(x1, w1s[64 + j],     b1s[j]));
        float f1 = fmaf(x0, w1s[j + 1], fmaf(x1, w1s[64 + j + 1], b1s[j + 1]));
        bx.u[p] = pack2(fmaxf(f0, 0.0f), fmaxf(f1, 0.0f));
      }
      Bx[kc] = bx.h;
    }

    // ---- layer 2: h2^T = relu(W2^T @ h1^T + b2); fused per-tile epilogue ----
    unsigned q2[4][4], s2[4][4];
#pragma unroll
    for (int mt = 0; mt < 4; ++mt) {
      v8f c = {};
      c = wmma16(A2[mt][0], Bx[0], c);
      c = wmma16(A2[mt][1], Bx[1], c);
      const v4f ba = *(const v4f*)(b2s + 16 * mt + 8 * hb);
      const v4f bb = *(const v4f*)(b2s + 16 * mt + 8 * hb + 4);
#pragma unroll
      for (int p = 0; p < 4; ++p) {
        float e0 = fmaxf(c[2 * p]     + ((p < 2) ? ba[2 * p]     : bb[2 * p - 4]), 0.0f);
        float e1 = fmaxf(c[2 * p + 1] + ((p < 2) ? ba[2 * p + 1] : bb[2 * p - 3]), 0.0f);
        q2[mt][p] = pack2(e0, e1);
      }
    }
    // cross-half block exchange (no LDS memory traffic)
#pragma unroll
    for (int mt = 0; mt < 4; ++mt)
#pragma unroll
      for (int p = 0; p < 4; ++p)
        s2[mt][p] = __shfl_xor((int)q2[mt][p], 16, 32);

    unsigned lo4[4], hi4[4];
    v16h B3[2];
#pragma unroll
    for (int p = 0; p < 4; ++p) {
      lo4[p] = hb ? s2[1][p] : q2[0][p];
      hi4[p] = hb ? q2[1][p] : s2[0][p];
    }
    B3[0] = assemble(lo4, hi4);
#pragma unroll
    for (int p = 0; p < 4; ++p) {
      lo4[p] = hb ? s2[3][p] : q2[2][p];
      hi4[p] = hb ? q2[3][p] : s2[2][p];
    }
    B3[1] = assemble(lo4, hi4);

    // ---- layer 3: A-fragments streamed from LDS, fused epilogue ----
    unsigned q3[4][4], s3[4][4];
#pragma unroll
    for (int mt = 0; mt < 4; ++mt) {
      const f16* r3 = w3t + (16 * mt + l15) * 64 + off;
      v16h a0 = load_frag(r3, 0, hb);
      v16h a1 = load_frag(r3, 1, hb);
      v8f c = {};
      c = wmma16(a0, B3[0], c);
      c = wmma16(a1, B3[1], c);
      const v4f ba = *(const v4f*)(b3s + 16 * mt + 8 * hb);
      const v4f bb = *(const v4f*)(b3s + 16 * mt + 8 * hb + 4);
#pragma unroll
      for (int p = 0; p < 4; ++p) {
        float e0 = fmaxf(c[2 * p]     + ((p < 2) ? ba[2 * p]     : bb[2 * p - 4]), 0.0f);
        float e1 = fmaxf(c[2 * p + 1] + ((p < 2) ? ba[2 * p + 1] : bb[2 * p - 3]), 0.0f);
        q3[mt][p] = pack2(e0, e1);
      }
    }
#pragma unroll
    for (int mt = 0; mt < 4; ++mt)
#pragma unroll
      for (int p = 0; p < 4; ++p)
        s3[mt][p] = __shfl_xor((int)q3[mt][p], 16, 32);

    v16h Bh[2];
#pragma unroll
    for (int p = 0; p < 4; ++p) {
      lo4[p] = hb ? s3[1][p] : q3[0][p];
      hi4[p] = hb ? q3[1][p] : s3[0][p];
    }
    Bh[0] = assemble(lo4, hi4);
#pragma unroll
    for (int p = 0; p < 4; ++p) {
      lo4[p] = hb ? s3[3][p] : q3[2][p];
      hi4[p] = hb ? q3[3][p] : s3[2][p];
    }
    Bh[1] = assemble(lo4, hi4);

    // ---- heads via WMMA: [mu(4) | lam(3) | pad]^T @ h3^T ----
    const f16* rh = whh + l15 * 64 + off;
    v8f ch = {};
    ch = wmma16(load_frag(rh, 0, hb), Bh[0], ch);
    ch = wmma16(load_frag(rh, 1, hb), Bh[1], ch);
    // Lane c (low half) VGPR v = head row v for point c.

    if (lane < 16 && active) {
      // mu: relu then clip ||(mu0-mu1, mu2-mu3)|| to <= 1
      float m0 = fmaxf(ch[0] + hbm[0], 0.0f);
      float m1 = fmaxf(ch[1] + hbm[1], 0.0f);
      float m2 = fmaxf(ch[2] + hbm[2], 0.0f);
      float m3 = fmaxf(ch[3] + hbm[3], 0.0f);
      const float ga = m0 - m1, gb = m2 - m3;
      const float nrm = sqrtf(ga * ga + gb * gb);
      const float sc = (nrm > 1.0f) ? (1.0f / (nrm + 1e-8f)) : 1.0f;
      out[0 * N + pt] = m0 * sc;
      out[1 * N + pt] = m1 * sc;
      out[2 * N + pt] = m2 * sc;
      out[3 * N + pt] = m3 * sc;
      // lam: normalize to unit norm
      float l0 = ch[4] + hbl[0];
      float l1 = ch[5] + hbl[1];
      float l2 = ch[6] + hbl[2];
      const float nl = sqrtf(fmaf(l0, l0, fmaf(l1, l1, l2 * l2)));
      const float inv = 1.0f / (nl + 1e-8f);
      out[4 * N + pt] = l0 * inv;
      out[5 * N + pt] = l1 * inv;
      out[6 * N + pt] = l2 * inv;
    }
  }
}

extern "C" void kernel_launch(void* const* d_in, const int* in_sizes, int n_in,
                              void* d_out, int out_size, void* d_ws, size_t ws_size,
                              hipStream_t stream) {
  (void)n_in; (void)d_ws; (void)ws_size; (void)out_size;
  const float* pc   = (const float*)d_in[0];
  const float* W1   = (const float*)d_in[1];
  const float* b1   = (const float*)d_in[2];
  const float* W2   = (const float*)d_in[3];
  const float* b2   = (const float*)d_in[4];
  const float* W3   = (const float*)d_in[5];
  const float* b3   = (const float*)d_in[6];
  const float* Wmu  = (const float*)d_in[7];
  const float* bmu  = (const float*)d_in[8];
  const float* Wlam = (const float*)d_in[9];
  const float* blam = (const float*)d_in[10];
  float* out = (float*)d_out;

  const int N = in_sizes[0] / 2;  // point_cloud is (N,2)
  const int ntiles  = (N + 15) >> 4;
  const int ngroups = (ntiles + WAVES_PER_BLOCK - 1) / WAVES_PER_BLOCK;
  const int blocks  = (ngroups < GRID_BLOCKS) ? ngroups : GRID_BLOCKS;

  mlp_hardproj_kernel<<<blocks, BLOCK_THREADS, 0, stream>>>(
      pc, W1, b1, W2, b2, W3, b3, Wmu, bmu, Wlam, blam, out, N);
}
